// MultiHeadAttentionWrapper_73495480369574
// MI455X (gfx1250) — compile-verified, matrix operand order checked by reading, not run on hardware
//
#include <hip/hip_runtime.h>
#include <hip/hip_bf16.h>

// MI455X (gfx1250) fused MHA: bf16 WMMA (f32 accum) flash attention.
// Phase 1: convert x->bf16, transpose+convert W -> [H,64,1024] bf16.
// Phase 2: QKV projection with v_wmma_f32_16x16x32_bf16 (12 WMMAs/K-step/wave).
// Phase 3: one-wave-per-query-tile flash attention (online softmax), causal.

typedef __attribute__((ext_vector_type(16))) __bf16 v16bf;
typedef __attribute__((ext_vector_type(8)))  __bf16 v8bf;
typedef __attribute__((ext_vector_type(8)))  float  v8f;

#define D_IN  1024
#define D_OUT 64
#define HEADS 16
#define SEQ   2048
#define BATCH 2

// Build a 16-bit A/B fragment (16x32 / 32x16) for this lane from a base
// pointer at the lane's first K element: elements {0..7} and {16..23}
// relative to that base (ISA 7.12.2 16-bit A-matrix layout).
__device__ __forceinline__ v16bf make_frag(const __bf16* p) {
  v8bf lo = *(const v8bf*)p;
  v8bf hi = *(const v8bf*)(p + 16);
  v16bf r;
#pragma unroll
  for (int i = 0; i < 8; ++i) { r[i] = lo[i]; r[i + 8] = hi[i]; }
  return r;
}

__device__ __forceinline__ v8f wmma_bf16(v16bf a, v16bf b, v8f c) {
  return __builtin_amdgcn_wmma_f32_16x16x32_bf16(
      /*neg_a=*/false, a, /*neg_b=*/false, b,
      /*c_mod=*/(short)0, c, /*reuse_a=*/false, /*reuse_b=*/false);
}

// ---------------------------------------------------------------------------
// Phase 1: x -> bf16 ; W[h][k][n] -> Wt[h][n][k] bf16 (contiguous B-fragments)
// ---------------------------------------------------------------------------
__global__ __launch_bounds__(256) void prep_kernel(
    const float* __restrict__ x,  const float* __restrict__ Wq,
    const float* __restrict__ Wk, const float* __restrict__ Wv,
    __bf16* __restrict__ xb, __bf16* __restrict__ wqt,
    __bf16* __restrict__ wkt, __bf16* __restrict__ wvt) {
  unsigned idx = blockIdx.x * 256u + threadIdx.x;
  if (idx < (unsigned)(BATCH * SEQ * D_IN)) xb[idx] = (__bf16)x[idx];
  if (idx < (unsigned)(HEADS * D_IN * D_OUT)) {
    unsigned h = idx >> 16;           // D_IN*D_OUT = 65536
    unsigned rem = idx & 65535u;
    unsigned k = rem >> 6;            // D_OUT = 64
    unsigned n = rem & 63u;
    unsigned dst = (h << 16) + (n << 10) + k;   // [h][n][k], k stride 1
    wqt[dst] = (__bf16)Wq[idx];
    wkt[dst] = (__bf16)Wk[idx];
    wvt[dst] = (__bf16)Wv[idx];
  }
}

// ---------------------------------------------------------------------------
// Phase 2: Q,K,V projection. One wave per (b,h,16-row tile).
// Q,K stored row-major bf16 [B,H,S,64]; V stored transposed [B,H,64,S].
// ---------------------------------------------------------------------------
__global__ __launch_bounds__(256) void qkv_kernel(
    const __bf16* __restrict__ xb,  const __bf16* __restrict__ wqt,
    const __bf16* __restrict__ wkt, const __bf16* __restrict__ wvt,
    __bf16* __restrict__ Q, __bf16* __restrict__ K, __bf16* __restrict__ Vt) {
  const int lane  = threadIdx.x & 31;
  const int gid   = blockIdx.x * 8 + (threadIdx.x >> 5);
  const int mt    = gid & 127;          // SEQ/16 tiles
  const int h     = (gid >> 7) & 15;
  const int b     = gid >> 11;
  const int l15   = lane & 15;
  const int half8 = (lane < 16) ? 0 : 8;

  const __bf16* xrow = xb + ((size_t)b * SEQ + mt * 16 + l15) * D_IN;
  const __bf16* wbase[3] = { wqt + (size_t)h * D_OUT * D_IN,
                             wkt + (size_t)h * D_OUT * D_IN,
                             wvt + (size_t)h * D_OUT * D_IN };

  v8f acc[3][4];
#pragma unroll
  for (int m3 = 0; m3 < 3; ++m3)
#pragma unroll
    for (int nt = 0; nt < 4; ++nt)
#pragma unroll
      for (int i = 0; i < 8; ++i) acc[m3][nt][i] = 0.0f;

  for (int kk = 0; kk < D_IN / 32; ++kk) {
    const int k0 = kk * 32 + half8;
    v16bf a = make_frag(xrow + k0);
#pragma unroll
    for (int m3 = 0; m3 < 3; ++m3) {
#pragma unroll
      for (int nt = 0; nt < 4; ++nt) {
        v16bf bf = make_frag(wbase[m3] + (size_t)(nt * 16 + l15) * D_IN + k0);
        acc[m3][nt] = wmma_bf16(a, bf, acc[m3][nt]);
      }
    }
  }

  const size_t bh = (size_t)(b * HEADS + h);
#pragma unroll
  for (int nt = 0; nt < 4; ++nt) {
    const int n = nt * 16 + l15;
#pragma unroll
    for (int i = 0; i < 8; ++i) {
      const int row = mt * 16 + half8 + i;
      const size_t qk = (bh * SEQ + row) * D_OUT + n;
      Q[qk] = (__bf16)acc[0][nt][i];
      K[qk] = (__bf16)acc[1][nt][i];
    }
    v8bf pk;
#pragma unroll
    for (int i = 0; i < 8; ++i) pk[i] = (__bf16)acc[2][nt][i];
    *(v8bf*)(Vt + (bh * D_OUT + n) * SEQ + mt * 16 + half8) = pk;  // packed 16B
  }
}

// ---------------------------------------------------------------------------
// Phase 3: causal flash attention. One wave per (b,h, 16-query tile).
// Key-pair tiles of 32 keys = one WMMA K-step for P@V.
// ---------------------------------------------------------------------------
__global__ __launch_bounds__(32) void flash_kernel(
    const __bf16* __restrict__ Q, const __bf16* __restrict__ K,
    const __bf16* __restrict__ Vt, float* __restrict__ out) {
  __shared__ __align__(16) __bf16 ldsP[16 * 32];
  const int lane  = threadIdx.x;
  const int l15   = lane & 15;
  const int half8 = (lane < 16) ? 0 : 8;
  const int qt    = blockIdx.x;
  const int bh    = blockIdx.y;
  const int b     = bh >> 4;
  const int h     = bh & 15;
  const size_t qkBase = (size_t)bh * SEQ * D_OUT;
  const size_t vtBase = (size_t)bh * D_OUT * SEQ;

  // Q A-fragments for both 32-wide K-steps over d=64 (loaded once).
  const __bf16* qrow = Q + qkBase + (size_t)(qt * 16 + l15) * D_OUT;
  v16bf aq[2];
#pragma unroll
  for (int s = 0; s < 2; ++s) aq[s] = make_frag(qrow + s * 32 + half8);

  v8f m8, l8, o[4];
#pragma unroll
  for (int i = 0; i < 8; ++i) { m8[i] = -1e30f; l8[i] = 0.0f; }
#pragma unroll
  for (int nt = 0; nt < 4; ++nt)
#pragma unroll
    for (int i = 0; i < 8; ++i) o[nt][i] = 0.0f;

  const int lastp = (qt * 16 + 15) >> 5;   // last key-pair tile (causal)
  const int rbase = qt * 16 + half8;       // this lane's row group base

  for (int kp = 0; kp <= lastp; ++kp) {
    // --- scores: two 16x16 tiles (keys 2kp*16 .. 2kp*16+31), K=64 ---
    v8f sc[2];
#pragma unroll
    for (int j = 0; j < 2; ++j) {
#pragma unroll
      for (int i = 0; i < 8; ++i) sc[j][i] = 0.0f;
      const __bf16* krow = K + qkBase + (size_t)((2 * kp + j) * 16 + l15) * D_OUT;
#pragma unroll
      for (int s = 0; s < 2; ++s)
        sc[j] = wmma_bf16(aq[s], make_frag(krow + s * 32 + half8), sc[j]);
    }
    // --- scale + causal mask (boundary tile only) ---
    const bool needMask = (kp == lastp);
#pragma unroll
    for (int j = 0; j < 2; ++j) {
      const int cglob = (2 * kp + j) * 16 + l15;
#pragma unroll
      for (int i = 0; i < 8; ++i) {
        float sv = sc[j][i] * 0.125f;                 // 1/sqrt(64)
        if (needMask && cglob > rbase + i) sv = -1e30f;
        sc[j][i] = sv;
      }
    }
    // --- online softmax: row max over 32 keys (butterfly in 16-lane half) ---
    v8f t;
#pragma unroll
    for (int i = 0; i < 8; ++i) t[i] = fmaxf(sc[0][i], sc[1][i]);
#pragma unroll
    for (int d = 1; d <= 8; d <<= 1)
#pragma unroll
      for (int i = 0; i < 8; ++i) t[i] = fmaxf(t[i], __shfl_xor(t[i], d, 32));

    v8f mnew, alpha, rs;
#pragma unroll
    for (int i = 0; i < 8; ++i) {
      mnew[i]  = fmaxf(m8[i], t[i]);
      alpha[i] = __expf(m8[i] - mnew[i]);
      sc[0][i] = __expf(sc[0][i] - mnew[i]);
      sc[1][i] = __expf(sc[1][i] - mnew[i]);
      rs[i]    = sc[0][i] + sc[1][i];
    }
#pragma unroll
    for (int d = 1; d <= 8; d <<= 1)
#pragma unroll
      for (int i = 0; i < 8; ++i) rs[i] += __shfl_xor(rs[i], d, 32);
#pragma unroll
    for (int i = 0; i < 8; ++i) {
      l8[i] = l8[i] * alpha[i] + rs[i];
      m8[i] = mnew[i];
    }
#pragma unroll
    for (int nt = 0; nt < 4; ++nt)
#pragma unroll
      for (int i = 0; i < 8; ++i) o[nt][i] *= alpha[i];

    // --- C/D-layout P (f32) -> A-layout bf16 via LDS (single-wave block) ---
#pragma unroll
    for (int j = 0; j < 2; ++j)
#pragma unroll
      for (int i = 0; i < 8; ++i)
        ldsP[(half8 + i) * 32 + j * 16 + l15] = (__bf16)sc[j][i];
    __syncthreads();
    v16bf ap;
    {
      v8bf lo = *(const v8bf*)(ldsP + l15 * 32 + half8);
      v8bf hi = *(const v8bf*)(ldsP + l15 * 32 + half8 + 16);
#pragma unroll
      for (int i = 0; i < 8; ++i) { ap[i] = lo[i]; ap[i + 8] = hi[i]; }
    }
    __syncthreads();

    // --- O += P @ V  (K=32 keys, N=64 features, V transposed => contiguous) ---
#pragma unroll
    for (int nt = 0; nt < 4; ++nt) {
      const __bf16* vp =
          Vt + vtBase + (size_t)(nt * 16 + l15) * SEQ + kp * 32 + half8;
      o[nt] = wmma_bf16(ap, make_frag(vp), o[nt]);
    }
  }

  // --- normalize and write [B, S, H*64] ---
  v8f rinv;
#pragma unroll
  for (int i = 0; i < 8; ++i) rinv[i] = 1.0f / l8[i];
#pragma unroll
  for (int nt = 0; nt < 4; ++nt) {
    const int n = h * D_OUT + nt * 16 + l15;
#pragma unroll
    for (int i = 0; i < 8; ++i) {
      const int row = qt * 16 + half8 + i;
      out[((size_t)b * SEQ + row) * (HEADS * D_OUT) + n] = o[nt][i] * rinv[i];
    }
  }
}

// ---------------------------------------------------------------------------
extern "C" void kernel_launch(void* const* d_in, const int* in_sizes, int n_in,
                              void* d_out, int out_size, void* d_ws, size_t ws_size,
                              hipStream_t stream) {
  (void)in_sizes; (void)n_in; (void)out_size; (void)ws_size;
  const float* x  = (const float*)d_in[0];
  const float* Wq = (const float*)d_in[1];
  const float* Wk = (const float*)d_in[2];
  const float* Wv = (const float*)d_in[3];
  float* out = (float*)d_out;

  const size_t NX = (size_t)BATCH * SEQ * D_IN;        // 4,194,304
  const size_t NW = (size_t)HEADS * D_IN * D_OUT;      // 1,048,576
  const size_t NQ = (size_t)BATCH * HEADS * SEQ * D_OUT; // 4,194,304

  __bf16* ws  = (__bf16*)d_ws;   // ~38 MB of bf16 staging (fits in L2)
  __bf16* xb  = ws;
  __bf16* wqt = xb + NX;
  __bf16* wkt = wqt + NW;
  __bf16* wvt = wkt + NW;
  __bf16* Qb  = wvt + NW;
  __bf16* Kb  = Qb + NQ;
  __bf16* Vt  = Kb + NQ;

  prep_kernel<<<(unsigned)((NX + 255) / 256), 256, 0, stream>>>(
      x, Wq, Wk, Wv, xb, wqt, wkt, wvt);

  // BATCH*HEADS*(SEQ/16) = 4096 waves, 8 waves (256 thr) per block
  qkv_kernel<<<BATCH * HEADS * (SEQ / 16) / 8, 256, 0, stream>>>(
      xb, wqt, wkt, wvt, Qb, Kb, Vt);

  flash_kernel<<<dim3(SEQ / 16, BATCH * HEADS), 32, 0, stream>>>(
      Qb, Kb, Vt, out);
}
